// LinearTimeCrossAttention_35562329211185
// MI455X (gfx1250) — compile-verified, matrix operand order checked by reading, not run on hardware
//
#include <hip/hip_runtime.h>
#include <hip/hip_bf16.h>

typedef _Float16 f16;
typedef __attribute__((ext_vector_type(16))) _Float16 v16h;
typedef __attribute__((ext_vector_type(8)))  float    v8f;
typedef __attribute__((ext_vector_type(4)))  unsigned int u32x4;
typedef __attribute__((ext_vector_type(2)))  unsigned int u32x2;
typedef int gvi4 __attribute__((vector_size(16)));   // matches async-LDS builtin param

union Frag16 { v16h h; u32x4 q[2]; };

#define AS1 __attribute__((address_space(1)))
#define AS3 __attribute__((address_space(3)))

#if __has_builtin(__builtin_amdgcn_global_load_async_to_lds_b128)
#define HAVE_ASYNC_LDS 1
#else
#define HAVE_ASYNC_LDS 0
#endif

__device__ __forceinline__ void wait_asynccnt0() {
#if __has_builtin(__builtin_amdgcn_s_wait_asynccnt)
    __builtin_amdgcn_s_wait_asynccnt(0);
#else
    asm volatile("s_wait_asynccnt 0x0" ::: "memory");
#endif
}

#define HEADS 4
#define DIMH 32
#define BATCH 16
#define CCH 256      // x channels
#define CD 512       // cond channels
#define HID 128      // heads*dim_head
#define NPIX 4096    // 64*64
#define NC 1024      // 32*32
#define GROUPS 32

// ---------------------------------------------------------------------------
// f32 -> f16 elementwise convert
// ---------------------------------------------------------------------------
__global__ __launch_bounds__(256) void cvt_f16_kernel(const float* __restrict__ src,
                                                      f16* __restrict__ dst, long n) {
    long i = (long)blockIdx.x * blockDim.x + threadIdx.x;
    long stride = (long)gridDim.x * blockDim.x;
    for (; i < n; i += stride) dst[i] = (f16)src[i];
}

// ---------------------------------------------------------------------------
// GroupNorm: x (B,256,4096) -> xn f16, 1 block per (batch, group)
// ---------------------------------------------------------------------------
__global__ __launch_bounds__(256) void groupnorm_kernel(const float* __restrict__ x,
                                                        const float* __restrict__ w,
                                                        const float* __restrict__ b,
                                                        f16* __restrict__ xn) {
    int blk = blockIdx.x;
    int bat = blk >> 5;
    int g   = blk & 31;
    const float* xb = x + ((long)bat * CCH + g * 8) * NPIX;
    f16*         ob = xn + ((long)bat * CCH + g * 8) * NPIX;
    int t = threadIdx.x;

    float s = 0.f, s2 = 0.f;
    for (int i = t; i < 8 * NPIX; i += 256) {
        float v = xb[i];
        s += v; s2 += v * v;
    }
    __shared__ float sh[256], sh2[256];
    sh[t] = s; sh2[t] = s2;
    __syncthreads();
    for (int o = 128; o > 0; o >>= 1) {
        if (t < o) { sh[t] += sh[t + o]; sh2[t] += sh2[t + o]; }
        __syncthreads();
    }
    float mu  = sh[0] * (1.0f / 32768.0f);
    float var = sh2[0] * (1.0f / 32768.0f) - mu * mu;
    float ri  = rsqrtf(var + 1e-5f);
    for (int i = t; i < 8 * NPIX; i += 256) {
        int ch = g * 8 + (i >> 12);
        float v = (xb[i] - mu) * ri;
        ob[i] = (f16)(v * w[ch] + b[ch]);
    }
}

// ---------------------------------------------------------------------------
// WMMA f16 GEMM:  C[b] = A[b] (MxK) * B[b] (KxN) + bias
//
// Block tile 64(M) x 64(N), BK=32, 8 waves; wave (w>>1, w&1) owns a 16x32
// output strip: one A fragment reused across two v_wmma_f32_16x16x32_f16.
// Double-buffered LDS, one barrier per K-step. A tile staged with
// GLOBAL_LOAD_ASYNC_TO_LDS_B128 (ASYNCcnt); B tile staged transposed [n][k]
// so fragments fill with two aligned 16B LDS loads per lane.
//
// Fragment layout (ISA 7.12.2, 16-bit 16x32): lane L (idx = L&15):
//   halves 0..7  = K (L<16 ? 0..7 : 8..15),  halves 8..15 = +16.
// ---------------------------------------------------------------------------
template <bool OUT16>
__global__ __launch_bounds__(256) void gemm_wmma_kernel(
    const f16* __restrict__ A, long sA,
    const f16* __restrict__ B, long sB,
    void* __restrict__ Cv, long sC,
    const float* __restrict__ bias,
    int M, int N, int K) {

    __shared__ f16 ldsA[2][64][32];   // [buf][m][k]
    __shared__ f16 ldsB[2][64][32];   // [buf][n][k]

    int bat = blockIdx.z;
    int m0  = blockIdx.x * 64;
    int n0  = blockIdx.y * 64;
    int t    = threadIdx.x;
    int lane = t & 31;
    int warp = t >> 5;
    int trow = (warp >> 1) * 16;     // 0,16,32,48
    int tcol = (warp & 1) * 32;      // 0,32
    int hs   = (lane >> 4) * 8;      // K-half select

    const f16* Ab = A + (long)bat * sA;
    const f16* Bb = B + (long)bat * sB;

    // staging coordinates: 2048 halves per tile, 8 per thread
    int ar_st = t >> 2;              // A row 0..63
    int ac_st = (t & 3) * 8;         // A k-col, multiple of 8
    int bk_st = (t * 8) >> 6;        // B k-row 0..31
    int bn_st = (t * 8) & 63;        // B n-col, multiple of 8

    v8f acc0 = {0.f, 0.f, 0.f, 0.f, 0.f, 0.f, 0.f, 0.f};
    v8f acc1 = {0.f, 0.f, 0.f, 0.f, 0.f, 0.f, 0.f, 0.f};

    int nk = K >> 5;
    u32x4 rb = *(const u32x4*)(Bb + (long)bk_st * N + n0 + bn_st);
#if HAVE_ASYNC_LDS
    __builtin_amdgcn_global_load_async_to_lds_b128(
        (AS1 gvi4*)(Ab + (long)(m0 + ar_st) * K + ac_st),
        (AS3 gvi4*)&ldsA[0][ar_st][ac_st], 0, 0);
#else
    u32x4 ra = *(const u32x4*)(Ab + (long)(m0 + ar_st) * K + ac_st);
#endif

    for (int kt = 0; kt < nk; kt++) {
        int buf = kt & 1;
        // ---- commit tile kt to LDS ---------------------------------------
        {
            f16 tmp[8];
            *(u32x4*)tmp = rb;
#pragma unroll
            for (int i = 0; i < 8; i++) ldsB[buf][bn_st + i][bk_st] = tmp[i];
        }
#if HAVE_ASYNC_LDS
        wait_asynccnt0();            // my async A chunk for tile kt landed
#else
        *(u32x4*)&ldsA[buf][ar_st][ac_st] = ra;
#endif
        __syncthreads();             // all waves' A+B chunks visible

        // ---- issue tile kt+1 (other buffer) ------------------------------
        if (kt + 1 < nk) {
            int k0 = (kt + 1) * 32;
            rb = *(const u32x4*)(Bb + (long)(k0 + bk_st) * N + n0 + bn_st);
#if HAVE_ASYNC_LDS
            __builtin_amdgcn_global_load_async_to_lds_b128(
                (AS1 gvi4*)(Ab + (long)(m0 + ar_st) * K + k0 + ac_st),
                (AS3 gvi4*)&ldsA[buf ^ 1][ar_st][ac_st], 0, 0);
#else
            ra = *(const u32x4*)(Ab + (long)(m0 + ar_st) * K + k0 + ac_st);
#endif
            if (kt + 2 < nk)         // prefetch B tile kt+2 -> global_prefetch_b8
                __builtin_prefetch((const void*)(Bb + (long)((kt + 2) * 32 + bk_st) * N
                                                 + n0 + bn_st), 0, 1);
        }

        // ---- fragments + 2x WMMA -----------------------------------------
        Frag16 fa, fb0, fb1;
        int arr = trow + (lane & 15);
        int bc  = tcol + (lane & 15);
        fa.q[0]  = *(const u32x4*)&ldsA[buf][arr][hs];
        fa.q[1]  = *(const u32x4*)&ldsA[buf][arr][16 + hs];
        fb0.q[0] = *(const u32x4*)&ldsB[buf][bc][hs];
        fb0.q[1] = *(const u32x4*)&ldsB[buf][bc][16 + hs];
        fb1.q[0] = *(const u32x4*)&ldsB[buf][bc + 16][hs];
        fb1.q[1] = *(const u32x4*)&ldsB[buf][bc + 16][16 + hs];
        acc0 = __builtin_amdgcn_wmma_f32_16x16x32_f16(
            false, fa.h, false, fb0.h, (short)0, acc0, false, false);
        acc1 = __builtin_amdgcn_wmma_f32_16x16x32_f16(
            false, fa.h, false, fb1.h, (short)0, acc1, false, false);
    }

    // ---- epilogue: VGPR r -> row (trow + r + 8*(lane>=16)) ----------------
    int col0  = n0 + tcol + (lane & 15);
    int rbase = m0 + trow + (lane >> 4) * 8;
    for (int r = 0; r < 8; r++) {
        int row = rbase + r;
        float bv = bias[row];
        float v0 = acc0[r] + bv;
        float v1 = acc1[r] + bv;
        long base = (long)bat * sC + (long)row * N;
        if (OUT16) {
            ((f16*)Cv)[base + col0]      = (f16)v0;
            ((f16*)Cv)[base + col0 + 16] = (f16)v1;
        } else {
            ((float*)Cv)[base + col0]      = v0;
            ((float*)Cv)[base + col0 + 16] = v1;
        }
    }
}

// ---------------------------------------------------------------------------
// Per-(batch, head): softmax(k, axis=n) then ctx[d][e] = sum_n p[d][n]*v[e][n]
// kv: (B, 256, 1024) f32;  k rows = h*32+d,  v rows = 128+h*32+e.
// ---------------------------------------------------------------------------
__global__ __launch_bounds__(256) void softmax_ctx_kernel(const float* __restrict__ kv,
                                                          float* __restrict__ ctx) {
    int h = blockIdx.x;
    int b = blockIdx.y;
    const float* kbase = kv + ((long)b * 256 + h * DIMH) * NC;
    const float* vbase = kv + ((long)b * 256 + HID + h * DIMH) * NC;

    __shared__ float rmax[DIMH], rinv[DIMH];
    __shared__ f16 ldsP[DIMH][128];
    __shared__ f16 ldsV[DIMH][128];

    int t = threadIdx.x;
    int lane = t & 31;
    int warp = t >> 5;

    // pass 1: per-row max and sum (each wave owns 4 rows)
    for (int rr = 0; rr < 4; rr++) {
        int row = warp * 4 + rr;
        const float* kr = kbase + (long)row * NC;
        float m = -1e30f;
        for (int i = lane; i < NC; i += 32) m = fmaxf(m, kr[i]);
        for (int o = 16; o >= 1; o >>= 1) m = fmaxf(m, __shfl_xor(m, o, 32));
        float s = 0.f;
        for (int i = lane; i < NC; i += 32) s += __expf(kr[i] - m);
        for (int o = 16; o >= 1; o >>= 1) s += __shfl_xor(s, o, 32);
        if (lane == 0) { rmax[row] = m; rinv[row] = 1.0f / s; }
    }
    __syncthreads();

    v8f acc = {0.f, 0.f, 0.f, 0.f, 0.f, 0.f, 0.f, 0.f};
    int trow = (warp >> 1) * 16;   // valid for warp<4
    int tcol = (warp & 1) * 16;
    int hs = (lane >> 4) * 8;

    for (int nc = 0; nc < NC; nc += 128) {
        int idx = t * 16;
        int d  = idx >> 7;        // 0..31
        int ns = idx & 127;
        float mr = rmax[d], sr = rinv[d];
        const float* kr = kbase + (long)d * NC + nc + ns;
        const float* vr = vbase + (long)d * NC + nc + ns;
        for (int i = 0; i < 16; i++) {
            ldsP[d][ns + i] = (f16)(__expf(kr[i] - mr) * sr);
            ldsV[d][ns + i] = (f16)vr[i];
        }
        __syncthreads();

        if (warp < 4) {
            int ar = trow + (lane & 15);
            int bc = tcol + (lane & 15);
            for (int kk = 0; kk < 128; kk += 32) {
                Frag16 fa, fb;
                fa.q[0] = *(const u32x4*)&ldsP[ar][kk + hs];
                fa.q[1] = *(const u32x4*)&ldsP[ar][kk + 16 + hs];
                fb.q[0] = *(const u32x4*)&ldsV[bc][kk + hs];
                fb.q[1] = *(const u32x4*)&ldsV[bc][kk + 16 + hs];
                acc = __builtin_amdgcn_wmma_f32_16x16x32_f16(
                    false, fa.h, false, fb.h, (short)0, acc, false, false);
            }
        }
        __syncthreads();
    }

    if (warp < 4) {
        float* cb = ctx + (((long)b * HEADS + h) * DIMH) * DIMH;
        int e = tcol + (lane & 15);
        for (int r = 0; r < 8; r++) {
            int d = trow + r + (lane >> 4) * 8;
            cb[d * DIMH + e] = acc[r];
        }
    }
}

// ---------------------------------------------------------------------------
// Fold context into output weight:
//   W2[b][c][h*32+d] = sum_e w_out[c][h*32+e] * ctx[b][h][d][e]   (f16 out)
// ---------------------------------------------------------------------------
__global__ __launch_bounds__(256) void fold_w2_kernel(const float* __restrict__ w_out,
                                                      const float* __restrict__ ctx,
                                                      f16* __restrict__ w2) {
    long o = (long)blockIdx.x * blockDim.x + threadIdx.x;   // B*256*128 elems
    if (o >= (long)BATCH * CCH * HID) return;
    int j = (int)(o & (HID - 1));
    long rem = o >> 7;
    int c = (int)(rem & (CCH - 1));
    int b = (int)(rem >> 8);
    int h = j >> 5;
    int d = j & 31;
    const float* wr = w_out + (long)c * HID + h * DIMH;
    const float* cr = ctx + (((long)b * HEADS + h) * DIMH + d) * DIMH;
    float acc = 0.f;
    for (int e = 0; e < DIMH; e++) acc += wr[e] * cr[e];
    w2[o] = (f16)acc;
}

// ---------------------------------------------------------------------------
extern "C" void kernel_launch(void* const* d_in, const int* in_sizes, int n_in,
                              void* d_out, int out_size, void* d_ws, size_t ws_size,
                              hipStream_t stream) {
    const float* x     = (const float*)d_in[0];   // (16,256,64,64)
    const float* cond  = (const float*)d_in[1];   // (16,512,32,32)
    const float* gn_w  = (const float*)d_in[2];
    const float* gn_b  = (const float*)d_in[3];
    const float* w_q   = (const float*)d_in[4];   // (128,256)
    const float* b_q   = (const float*)d_in[5];
    const float* w_kv  = (const float*)d_in[6];   // (256,512)
    const float* b_kv  = (const float*)d_in[7];
    const float* w_out = (const float*)d_in[8];   // (256,128)
    const float* b_out = (const float*)d_in[9];
    float* out = (float*)d_out;                   // (16,256,64,64)

    // workspace layout
    char* ws = (char*)d_ws;
    constexpr long SZ_XN   = (long)BATCH * CCH * NPIX * 2;   // 33.5 MB f16
    constexpr long SZ_CD16 = (long)BATCH * CD * NC * 2;      // 16.8 MB f16
    constexpr long SZ_WQ   = (long)HID * CCH * 2;
    constexpr long SZ_WKV  = (long)2 * HID * CD * 2;
    constexpr long SZ_KV   = (long)BATCH * 2 * HID * NC * 4; // f32
    constexpr long SZ_Q16  = (long)BATCH * HID * NPIX * 2;
    constexpr long SZ_CTX  = (long)BATCH * HEADS * DIMH * DIMH * 4;
    f16*   xn16   = (f16*)(ws);
    f16*   cond16 = (f16*)(ws + SZ_XN);
    f16*   wq16   = (f16*)(ws + SZ_XN + SZ_CD16);
    f16*   wkv16  = (f16*)(ws + SZ_XN + SZ_CD16 + SZ_WQ);
    float* kvf    = (float*)(ws + SZ_XN + SZ_CD16 + SZ_WQ + SZ_WKV);
    f16*   q16    = (f16*)(ws + SZ_XN + SZ_CD16 + SZ_WQ + SZ_WKV + SZ_KV);
    float* ctxf   = (float*)(ws + SZ_XN + SZ_CD16 + SZ_WQ + SZ_WKV + SZ_KV + SZ_Q16);
    f16*   w2_16  = (f16*)(ws + SZ_XN + SZ_CD16 + SZ_WQ + SZ_WKV + SZ_KV + SZ_Q16 + SZ_CTX);

    // 1) converts
    long ncond = (long)BATCH * CD * NC;
    cvt_f16_kernel<<<(ncond + 255) / 256, 256, 0, stream>>>(cond, cond16, ncond);
    cvt_f16_kernel<<<(HID * CCH + 255) / 256, 256, 0, stream>>>(w_q, wq16, HID * CCH);
    cvt_f16_kernel<<<(2 * HID * CD + 255) / 256, 256, 0, stream>>>(w_kv, wkv16, 2 * HID * CD);

    // 2) GroupNorm -> xn f16
    groupnorm_kernel<<<BATCH * GROUPS, 256, 0, stream>>>(x, gn_w, gn_b, xn16);

    // 3) kv = w_kv @ cond + b_kv   (M=256,K=512,N=1024, f32 out)
    gemm_wmma_kernel<false><<<dim3(256 / 64, NC / 64, BATCH), 256, 0, stream>>>(
        wkv16, 0, cond16, (long)CD * NC, kvf, (long)2 * HID * NC, b_kv, 2 * HID, NC, CD);

    // 4) q = w_q @ xn + b_q        (M=128,K=256,N=4096, f16 out)
    gemm_wmma_kernel<true><<<dim3(HID / 64, NPIX / 64, BATCH), 256, 0, stream>>>(
        wq16, 0, xn16, (long)CCH * NPIX, q16, (long)HID * NPIX, b_q, HID, NPIX, CCH);

    // 5) softmax(k) + context = P @ V^T  per (b,h)
    softmax_ctx_kernel<<<dim3(HEADS, BATCH), 256, 0, stream>>>(kvf, ctxf);

    // 6) fold ctx into output weight: W2 = w_out x ctx
    long nw2 = (long)BATCH * CCH * HID;
    fold_w2_kernel<<<(nw2 + 255) / 256, 256, 0, stream>>>(w_out, ctxf, w2_16);

    // 7) out = W2 @ q + b_out      (M=256,K=128,N=4096, f32 out)
    gemm_wmma_kernel<false><<<dim3(CCH / 64, NPIX / 64, BATCH), 256, 0, stream>>>(
        w2_16, (long)CCH * HID, q16, (long)HID * NPIX, out, (long)CCH * NPIX, b_out,
        CCH, NPIX, HID);
}
